// ResConv_29411936043448
// MI455X (gfx1250) — compile-verified, matrix operand order checked by reading, not run on hardware
//
#include <hip/hip_runtime.h>
#include <hip/hip_bf16.h>

typedef __attribute__((ext_vector_type(16))) __bf16 v16bf;
typedef __attribute__((ext_vector_type(4)))  __bf16 v4bf;
typedef __attribute__((ext_vector_type(8)))  float  v8f;

#define BVT      4
#define S_NODES  12288
#define FIN      128
#define FOUT     256
#define KNB      9
#define NGROUPS  32
#define GN_EPS   1e-5f

// LDS row padding (in bf16 elems): keep 32B alignment, 2-way max bank conflict
#define AROW1    144   // conv1 chunk row: 128 data + 16 pad  (288B, 72 dwords)
#define AROW2    272   // conv2 chunk row: 256 data + 16 pad  (544B, 136 dwords)

static __device__ __forceinline__ __bf16 f2bf(float f) {
    union { float f; unsigned u; } v; v.f = f;
    unsigned r = v.u + 0x7FFFu + ((v.u >> 16) & 1u);   // round-to-nearest-even
    unsigned short h = (unsigned short)(r >> 16);
    return __builtin_bit_cast(__bf16, h);
}

static __device__ __forceinline__ float silu(float x) {
    return x / (1.0f + __expf(-x));
}

// ---------------------------------------------------------------------------
// Weight prep: transpose + convert to bf16, layout [O][K*F] so B-fragments are
// 32 contiguous bytes per lane.
// ---------------------------------------------------------------------------
__global__ void prep_weights(const float* __restrict__ W1,
                             const float* __restrict__ W2,
                             const float* __restrict__ Wsk,
                             __bf16* __restrict__ W1t,
                             __bf16* __restrict__ W2t,
                             __bf16* __restrict__ Wst) {
    int i = blockIdx.x * blockDim.x + threadIdx.x;
    if (i < FOUT * KNB * FOUT) {                 // 589824: W2t[o][k*256+f]
        int o = i / (KNB * FOUT), kk = i % (KNB * FOUT);
        int k = kk / FOUT, f = kk % FOUT;
        W2t[i] = f2bf(W2[((size_t)k * FOUT + f) * FOUT + o]);
    }
    if (i < FOUT * KNB * FIN) {                  // 294912: W1t[o][k*128+f]
        int o = i / (KNB * FIN), kk = i % (KNB * FIN);
        int k = kk / FIN, f = kk % FIN;
        W1t[i] = f2bf(W1[((size_t)k * FIN + f) * FOUT + o]);
    }
    if (i < FOUT * FIN) {                        // 32768: Wst[o][f]
        int o = i / FIN, f = i % FIN;
        Wst[i] = f2bf(Wsk[(size_t)f * FOUT + o]);
    }
}

// ---------------------------------------------------------------------------
// GroupNorm stats: one block per (bvt, group); reduce over S * CPG elements.
// ---------------------------------------------------------------------------
template <int CPG, int F>
__global__ void gn_stats(const float* __restrict__ x, float2* __restrict__ stats) {
    const int bvt = blockIdx.x / NGROUPS;
    const int g   = blockIdx.x % NGROUPS;
    const float* base = x + (size_t)bvt * S_NODES * F + g * CPG;
    float sum = 0.f, sq = 0.f;
    for (int s = threadIdx.x; s < S_NODES; s += blockDim.x) {
        const float* p = base + (size_t)s * F;
#pragma unroll
        for (int c = 0; c < CPG; ++c) { float v = p[c]; sum += v; sq += v * v; }
    }
    __shared__ float s1[256], s2[256];
    s1[threadIdx.x] = sum; s2[threadIdx.x] = sq;
    __syncthreads();
    for (int off = 128; off > 0; off >>= 1) {
        if ((int)threadIdx.x < off) {
            s1[threadIdx.x] += s1[threadIdx.x + off];
            s2[threadIdx.x] += s2[threadIdx.x + off];
        }
        __syncthreads();
    }
    if (threadIdx.x == 0) {
        const float n = (float)(S_NODES * CPG);
        float mean = s1[0] / n;
        float var  = s2[0] / n - mean * mean;
        stats[blockIdx.x] = make_float2(mean, rsqrtf(var + GN_EPS));
    }
}

// ---------------------------------------------------------------------------
// Apply GN1 + SiLU -> h1 (bf16); also emit bf16 copy of x for the skip GEMM.
// ---------------------------------------------------------------------------
__global__ void gn1_apply(const float* __restrict__ x, const float2* __restrict__ st,
                          const float* __restrict__ sc, const float* __restrict__ bi,
                          __bf16* __restrict__ h1, __bf16* __restrict__ xb) {
    int i4  = blockIdx.x * blockDim.x + threadIdx.x;
    int f4  = i4 & 31;                 // group id (CPG=4)
    int row = i4 >> 5;
    int bvt = row / S_NODES;
    float2 ms = st[bvt * NGROUPS + f4];
    float4 xv = ((const float4*)x)[i4];
    int f0 = f4 << 2;
    float vals[4] = { xv.x, xv.y, xv.z, xv.w };
    v4bf hv, xv16;
#pragma unroll
    for (int c = 0; c < 4; ++c) {
        float xn = (vals[c] - ms.x) * ms.y * sc[f0 + c] + bi[f0 + c];
        hv[c]   = f2bf(silu(xn));
        xv16[c] = f2bf(vals[c]);
    }
    *(v4bf*)(h1 + (size_t)i4 * 4) = hv;
    *(v4bf*)(xb + (size_t)i4 * 4) = xv16;
}

// ---------------------------------------------------------------------------
// Apply GN2 + SiLU -> h2 (bf16). F=256, CPG=8 -> group = (float4 idx)/2.
// ---------------------------------------------------------------------------
__global__ void gn2_apply(const float* __restrict__ hp, const float2* __restrict__ st,
                          const float* __restrict__ sc, const float* __restrict__ bi,
                          __bf16* __restrict__ h2) {
    int i4  = blockIdx.x * blockDim.x + threadIdx.x;
    int f4  = i4 & 63;
    int row = i4 >> 6;
    int bvt = row / S_NODES;
    float2 ms = st[bvt * NGROUPS + (f4 >> 1)];
    float4 xv = ((const float4*)hp)[i4];
    int f0 = f4 << 2;
    float vals[4] = { xv.x, xv.y, xv.z, xv.w };
    v4bf hv;
#pragma unroll
    for (int c = 0; c < 4; ++c) {
        float xn = (vals[c] - ms.x) * ms.y * sc[f0 + c] + bi[f0 + c];
        hv[c] = f2bf(silu(xn));
    }
    *(v4bf*)(h2 + (size_t)i4 * 4) = hv;
}

// ---------------------------------------------------------------------------
// conv1: block = 64 nodes x 256 outputs (768 blocks). 8 waves; each wave owns
// 4 M-tiles x 2 N-tiles (C = 64 VGPRs) -> 0.75 operand fragments per WMMA.
// Gathered A chunk (1 neighbor x 64 nodes x 128) staged in LDS per k-chunk.
// ---------------------------------------------------------------------------
__global__ void __launch_bounds__(256)
conv1_wmma(const __bf16* __restrict__ h1, const int* __restrict__ adjc,
           const __bf16* __restrict__ W1t, const float* __restrict__ b1,
           float* __restrict__ h2pre) {
    __shared__ __bf16 As[64 * AROW1];            // 18.4 KB

    const int tile  = blockIdx.x % (S_NODES / 64);
    const int bvt   = blockIdx.x / (S_NODES / 64);
    const int baseS = tile * 64;
    const int wave  = threadIdx.x >> 5;
    const int lane  = threadIdx.x & 31;
    const int row   = lane & 15;
    const int khalf = lane >> 4;
    const int o0    = wave * 32 + row;           // B column (output channel)

    const __bf16* hb = h1 + (size_t)bvt * S_NODES * FIN;
    const int fill_node = threadIdx.x >> 2;      // 64 nodes
    const int fill_q    = threadIdx.x & 3;       // 4 x 32 bf16 per row

    v8f c[8];
#pragma unroll
    for (int t = 0; t < 8; ++t) c[t] = (v8f){};

    for (int kn = 0; kn < KNB; ++kn) {
        __syncthreads();
        // cooperative gather: one neighbor column of the 64-node tile
        {
            const int nb = adjc[(baseS + fill_node) * KNB + kn];
            const __bf16* src = hb + (size_t)nb * FIN + fill_q * 32;
            __bf16* dst = &As[fill_node * AROW1 + fill_q * 32];
            v16bf t0 = *(const v16bf*)(src);
            v16bf t1 = *(const v16bf*)(src + 16);
            *(v16bf*)(dst)      = t0;
            *(v16bf*)(dst + 16) = t1;
        }
        __syncthreads();

#pragma unroll
        for (int kc = 0; kc < FIN / 32; ++kc) {
            const int col = kc * 32 + khalf * 16;
            const int gk  = kn * FIN + col;
            v16bf a0 = *(const v16bf*)&As[( 0 + row) * AROW1 + col];
            v16bf a1 = *(const v16bf*)&As[(16 + row) * AROW1 + col];
            v16bf a2 = *(const v16bf*)&As[(32 + row) * AROW1 + col];
            v16bf a3 = *(const v16bf*)&As[(48 + row) * AROW1 + col];
            v16bf bA = *(const v16bf*)(W1t + (size_t)o0 * (KNB * FIN) + gk);
            v16bf bB = *(const v16bf*)(W1t + (size_t)(o0 + 16) * (KNB * FIN) + gk);
            c[0] = __builtin_amdgcn_wmma_f32_16x16x32_bf16(false, a0, false, bA, (short)0, c[0], false, false);
            c[1] = __builtin_amdgcn_wmma_f32_16x16x32_bf16(false, a0, false, bB, (short)0, c[1], false, false);
            c[2] = __builtin_amdgcn_wmma_f32_16x16x32_bf16(false, a1, false, bA, (short)0, c[2], false, false);
            c[3] = __builtin_amdgcn_wmma_f32_16x16x32_bf16(false, a1, false, bB, (short)0, c[3], false, false);
            c[4] = __builtin_amdgcn_wmma_f32_16x16x32_bf16(false, a2, false, bA, (short)0, c[4], false, false);
            c[5] = __builtin_amdgcn_wmma_f32_16x16x32_bf16(false, a2, false, bB, (short)0, c[5], false, false);
            c[6] = __builtin_amdgcn_wmma_f32_16x16x32_bf16(false, a3, false, bA, (short)0, c[6], false, false);
            c[7] = __builtin_amdgcn_wmma_f32_16x16x32_bf16(false, a3, false, bB, (short)0, c[7], false, false);
        }
    }

    float* obase = h2pre + ((size_t)bvt * S_NODES + baseS) * FOUT;
    const float bias0 = b1[o0];
    const float bias1 = b1[o0 + 16];
#pragma unroll
    for (int mt = 0; mt < 4; ++mt) {
#pragma unroll
        for (int r = 0; r < 8; ++r) {
            const int m = mt * 16 + r + khalf * 8;
            obase[(size_t)m * FOUT + o0]      = c[mt * 2 + 0][r] + bias0;
            obase[(size_t)m * FOUT + o0 + 16] = c[mt * 2 + 1][r] + bias1;
        }
    }
}

// ---------------------------------------------------------------------------
// conv2 + fused skip GEMM + biases -> final output. Same 64x256 block tiling.
// ---------------------------------------------------------------------------
__global__ void __launch_bounds__(256)
conv2_wmma(const __bf16* __restrict__ h2, const __bf16* __restrict__ xb,
           const int* __restrict__ adjc,
           const __bf16* __restrict__ W2t, const __bf16* __restrict__ Wst,
           const float* __restrict__ b2, const float* __restrict__ bsk,
           float* __restrict__ out) {
    __shared__ __bf16 As[64 * AROW2];            // 34.8 KB

    const int tile  = blockIdx.x % (S_NODES / 64);
    const int bvt   = blockIdx.x / (S_NODES / 64);
    const int baseS = tile * 64;
    const int wave  = threadIdx.x >> 5;
    const int lane  = threadIdx.x & 31;
    const int row   = lane & 15;
    const int khalf = lane >> 4;
    const int o0    = wave * 32 + row;

    const __bf16* hb = h2 + (size_t)bvt * S_NODES * FOUT;
    const int fill_node = threadIdx.x >> 2;
    const int fill_q    = threadIdx.x & 3;       // 4 x 64 bf16 per row

    v8f c[8];
#pragma unroll
    for (int t = 0; t < 8; ++t) c[t] = (v8f){};

    for (int kn = 0; kn < KNB; ++kn) {
        __syncthreads();
        {
            const int nb = adjc[(baseS + fill_node) * KNB + kn];
            const __bf16* src = hb + (size_t)nb * FOUT + fill_q * 64;
            __bf16* dst = &As[fill_node * AROW2 + fill_q * 64];
#pragma unroll
            for (int u = 0; u < 4; ++u)
                *(v16bf*)(dst + u * 16) = *(const v16bf*)(src + u * 16);
        }
        __syncthreads();

#pragma unroll
        for (int kc = 0; kc < FOUT / 32; ++kc) {
            const int col = kc * 32 + khalf * 16;
            const int gk  = kn * FOUT + col;
            v16bf a0 = *(const v16bf*)&As[( 0 + row) * AROW2 + col];
            v16bf a1 = *(const v16bf*)&As[(16 + row) * AROW2 + col];
            v16bf a2 = *(const v16bf*)&As[(32 + row) * AROW2 + col];
            v16bf a3 = *(const v16bf*)&As[(48 + row) * AROW2 + col];
            v16bf bA = *(const v16bf*)(W2t + (size_t)o0 * (KNB * FOUT) + gk);
            v16bf bB = *(const v16bf*)(W2t + (size_t)(o0 + 16) * (KNB * FOUT) + gk);
            c[0] = __builtin_amdgcn_wmma_f32_16x16x32_bf16(false, a0, false, bA, (short)0, c[0], false, false);
            c[1] = __builtin_amdgcn_wmma_f32_16x16x32_bf16(false, a0, false, bB, (short)0, c[1], false, false);
            c[2] = __builtin_amdgcn_wmma_f32_16x16x32_bf16(false, a1, false, bA, (short)0, c[2], false, false);
            c[3] = __builtin_amdgcn_wmma_f32_16x16x32_bf16(false, a1, false, bB, (short)0, c[3], false, false);
            c[4] = __builtin_amdgcn_wmma_f32_16x16x32_bf16(false, a2, false, bA, (short)0, c[4], false, false);
            c[5] = __builtin_amdgcn_wmma_f32_16x16x32_bf16(false, a2, false, bB, (short)0, c[5], false, false);
            c[6] = __builtin_amdgcn_wmma_f32_16x16x32_bf16(false, a3, false, bA, (short)0, c[6], false, false);
            c[7] = __builtin_amdgcn_wmma_f32_16x16x32_bf16(false, a3, false, bB, (short)0, c[7], false, false);
        }
    }

    // skip path: x(bf16) @ W_skip^T accumulated into the same tiles (K = 128)
    const __bf16* xbase = xb + ((size_t)bvt * S_NODES + baseS) * FIN;
#pragma unroll
    for (int kc = 0; kc < FIN / 32; ++kc) {
        const int col = kc * 32 + khalf * 16;
        v16bf a0 = *(const v16bf*)(xbase + (size_t)( 0 + row) * FIN + col);
        v16bf a1 = *(const v16bf*)(xbase + (size_t)(16 + row) * FIN + col);
        v16bf a2 = *(const v16bf*)(xbase + (size_t)(32 + row) * FIN + col);
        v16bf a3 = *(const v16bf*)(xbase + (size_t)(48 + row) * FIN + col);
        v16bf bA = *(const v16bf*)(Wst + (size_t)o0 * FIN + col);
        v16bf bB = *(const v16bf*)(Wst + (size_t)(o0 + 16) * FIN + col);
        c[0] = __builtin_amdgcn_wmma_f32_16x16x32_bf16(false, a0, false, bA, (short)0, c[0], false, false);
        c[1] = __builtin_amdgcn_wmma_f32_16x16x32_bf16(false, a0, false, bB, (short)0, c[1], false, false);
        c[2] = __builtin_amdgcn_wmma_f32_16x16x32_bf16(false, a1, false, bA, (short)0, c[2], false, false);
        c[3] = __builtin_amdgcn_wmma_f32_16x16x32_bf16(false, a1, false, bB, (short)0, c[3], false, false);
        c[4] = __builtin_amdgcn_wmma_f32_16x16x32_bf16(false, a2, false, bA, (short)0, c[4], false, false);
        c[5] = __builtin_amdgcn_wmma_f32_16x16x32_bf16(false, a2, false, bB, (short)0, c[5], false, false);
        c[6] = __builtin_amdgcn_wmma_f32_16x16x32_bf16(false, a3, false, bA, (short)0, c[6], false, false);
        c[7] = __builtin_amdgcn_wmma_f32_16x16x32_bf16(false, a3, false, bB, (short)0, c[7], false, false);
    }

    float* obase = out + ((size_t)bvt * S_NODES + baseS) * FOUT;
    const float bias0 = b2[o0]      + bsk[o0];
    const float bias1 = b2[o0 + 16] + bsk[o0 + 16];
#pragma unroll
    for (int mt = 0; mt < 4; ++mt) {
#pragma unroll
        for (int r = 0; r < 8; ++r) {
            const int m = mt * 16 + r + khalf * 8;
            obase[(size_t)m * FOUT + o0]      = c[mt * 2 + 0][r] + bias0;
            obase[(size_t)m * FOUT + o0 + 16] = c[mt * 2 + 1][r] + bias1;
        }
    }
}

// ---------------------------------------------------------------------------
extern "C" void kernel_launch(void* const* d_in, const int* in_sizes, int n_in,
                              void* d_out, int out_size, void* d_ws, size_t ws_size,
                              hipStream_t stream) {
    // setup_inputs() order:
    const float* x      = (const float*)d_in[0];
    const int*   adjc   = (const int*)  d_in[1];
    const float* gn1_sc = (const float*)d_in[2];
    const float* gn1_bi = (const float*)d_in[3];
    const float* gn2_sc = (const float*)d_in[4];
    const float* gn2_bi = (const float*)d_in[5];
    const float* W_skip = (const float*)d_in[6];
    const float* b_skip = (const float*)d_in[7];
    const float* W1     = (const float*)d_in[8];
    const float* b1     = (const float*)d_in[9];
    const float* W2     = (const float*)d_in[10];
    const float* b2     = (const float*)d_in[11];
    float* out = (float*)d_out;

    char* ws = (char*)d_ws;
    size_t off = 0;
    auto take = [&](size_t bytes) -> char* {
        char* p = ws + off;
        off += (bytes + 255) & ~(size_t)255;
        return p;
    };
    __bf16* W1t   = (__bf16*)take((size_t)FOUT * KNB * FIN  * 2);     // 0.59 MB
    __bf16* W2t   = (__bf16*)take((size_t)FOUT * KNB * FOUT * 2);     // 1.18 MB
    __bf16* Wst   = (__bf16*)take((size_t)FOUT * FIN * 2);            // 64 KB
    __bf16* h1    = (__bf16*)take((size_t)BVT * S_NODES * FIN  * 2);  // 12.6 MB
    __bf16* xb    = (__bf16*)take((size_t)BVT * S_NODES * FIN  * 2);  // 12.6 MB
    float*  h2pre = (float*) take((size_t)BVT * S_NODES * FOUT * 4);  // 50.3 MB
    __bf16* h2    = (__bf16*)take((size_t)BVT * S_NODES * FOUT * 2);  // 25.2 MB
    float2* st1   = (float2*)take((size_t)BVT * NGROUPS * sizeof(float2));
    float2* st2   = (float2*)take((size_t)BVT * NGROUPS * sizeof(float2));
    (void)ws_size; (void)in_sizes; (void)n_in; (void)out_size;

    const int conv_blocks = BVT * (S_NODES / 64);       // 768

    prep_weights<<<(FOUT * KNB * FOUT + 255) / 256, 256, 0, stream>>>(W1, W2, W_skip, W1t, W2t, Wst);

    gn_stats<4, FIN><<<BVT * NGROUPS, 256, 0, stream>>>(x, st1);
    gn1_apply<<<(BVT * S_NODES * FIN / 4) / 256, 256, 0, stream>>>(x, st1, gn1_sc, gn1_bi, h1, xb);

    conv1_wmma<<<conv_blocks, 256, 0, stream>>>(h1, adjc, W1t, b1, h2pre);

    gn_stats<8, FOUT><<<BVT * NGROUPS, 256, 0, stream>>>(h2pre, st2);
    gn2_apply<<<(BVT * S_NODES * FOUT / 4) / 256, 256, 0, stream>>>(h2pre, st2, gn2_sc, gn2_bi, h2);

    conv2_wmma<<<conv_blocks, 256, 0, stream>>>(h2, xb, adjc, W2t, Wst, b2, b_skip, out);
}
